// Transformer_52467320488536
// MI455X (gfx1250) — matrix operand order repaired and checked by reading, not compiled, not run on hardware
//
#include <hip/hip_runtime.h>

// ---------------------------------------------------------------------------
// Transformer forward (encoder-decoder) for MI455X / gfx1250, wave32 + WMMA.
// All matmuls: V_WMMA_F32_16X16X32_BF16 (bf16 A/B, f32 accumulate).
// ---------------------------------------------------------------------------

typedef __attribute__((ext_vector_type(16))) __bf16 v16bf;
typedef __attribute__((ext_vector_type(8)))  __bf16 v8bf;
typedef __attribute__((ext_vector_type(8)))  float  v8f;

#define BATCH 4
#define SEQ   512
#define DM    512
#define NH    8
#define HD    64
#define FF    2048
#define VOC   32000
#define NL    6
#define ATT_SCALE 0.125f   // 1/sqrt(64)

#define BM 128
#define BK 32

// ---------------------------------------------------------------------------
// Batched strided GEMM: C[z](m,n) = sum_k A[z](m,k) * B[z](k,n)  (+bias,+relu)
//   A(m,k) at A + zb*oAb + zh*oAh + m*lda + k          (k contiguous)
//   B(k,n) at B + zb*oBb + zh*oBh + k*sBk + n*sBn
//   C(m,n) at C + zb*oCb + zh*oCh + m*ldc + n
//
// BPATH=0 (sBn==1): B staged k-QUAD interleaved Bs[BK/4][BNT*4+16]: column c
//   holds k=4j..4j+3 contiguously -> fragment VGPR pair = one ds_load_b64.
//   +16 element pad puts the 4-row (lane16-31) offset 32 banks away.
// BPATH=1 (sBk==1): B staged transposed Bs[BNT][BK+8]; fragment = 2 x b128.
// BNT in {64,128}: 8 waves as 4(M) x 2(N); wave computes 32 x (BNT/2).
// Ping-pong LDS double buffer: one barrier per k-iteration.
// Caller guarantees exact tiling and 16B alignment. No bounds checks.
// ---------------------------------------------------------------------------
template <int BPATH, int BNT>
__global__ __launch_bounds__(256) void gemm_wmma_bf16(
    const float* __restrict__ A, const float* __restrict__ Bm,
    const float* __restrict__ bias, float* __restrict__ C,
    int M, int N, int K,
    long lda, long sBk, long sBn, long ldc,
    int Hb,
    long oAb, long oAh, long oBb, long oBh, long oCb, long oCh,
    int relu)
{
  constexpr int NSUB = BNT / 32;                 // wmma col subtiles per wave
  constexpr int G    = BNT / 32;                 // path0 cols per staging chunk
  constexpr int BSR  = (BPATH == 0) ? (BK / 4) : BNT;
  constexpr int BSC  = (BPATH == 0) ? (BNT * 4 + 16) : (BK + 8);
  constexpr int NB1  = BNT / 64;                 // path1 staging chunks/thread

  __shared__ __bf16 As[2][BM][BK + 8];           // row stride 80B (16B mult)
  __shared__ __bf16 Bs[2][BSR][BSC];

  const int zb = blockIdx.z / Hb;
  const int zh = blockIdx.z % Hb;
  const float* Ab = A  + (long)zb * oAb + (long)zh * oAh;
  const float* Bb = Bm + (long)zb * oBb + (long)zh * oBh;
  float*       Cb = C  + (long)zb * oCb + (long)zh * oCh;

  const int m0   = blockIdx.y * BM;
  const int n0   = blockIdx.x * BNT;
  const int tid  = threadIdx.x;
  const int lane = tid & 31;
  const int wid  = tid >> 5;
  const int wm   = wid & 3;    // 4 wave-rows x 32 = 128
  const int wn   = wid >> 2;   // 2 wave-cols x (BNT/2)

  const v8f vzero = {0.f, 0.f, 0.f, 0.f, 0.f, 0.f, 0.f, 0.f};
  v8f acc[2][NSUB];
#pragma unroll
  for (int mi = 0; mi < 2; ++mi)
#pragma unroll
    for (int ni = 0; ni < NSUB; ++ni) acc[mi][ni] = vzero;

  // --- register double buffers (global -> regs -> LDS pipeline) -------------
  float4 aR[2][2];       // A: 512 dbl-chunks (8 floats) / 256 thr = 2
  float4 bq[4];          // path0: 4 k-rows x G cols (G<=4)
  float4 b1R[NB1][2];    // path1 chunks

  auto ldA = [&](int kk) {
#pragma unroll
    for (int i = 0; i < 2; ++i) {
      const int e  = tid + i * 256;        // [0,512)
      const int r  = e >> 2;               // 4 dbl-chunks per 32-float row
      const int c8 = (e & 3) << 3;
      const float* p = Ab + (long)(m0 + r) * lda + kk + c8;
      aR[i][0] = *reinterpret_cast<const float4*>(p);
      aR[i][1] = *reinterpret_cast<const float4*>(p + 4);
    }
  };
  auto stA = [&](int buf) {
#pragma unroll
    for (int i = 0; i < 2; ++i) {
      const int e  = tid + i * 256;
      const int r  = e >> 2;
      const int c8 = (e & 3) << 3;
      v8bf o;
      o[0] = (__bf16)aR[i][0].x; o[1] = (__bf16)aR[i][0].y;
      o[2] = (__bf16)aR[i][0].z; o[3] = (__bf16)aR[i][0].w;
      o[4] = (__bf16)aR[i][1].x; o[5] = (__bf16)aR[i][1].y;
      o[6] = (__bf16)aR[i][1].z; o[7] = (__bf16)aR[i][1].w;
      *reinterpret_cast<v8bf*>(&As[buf][r][c8]) = o;   // one ds_store_b128
    }
  };
  auto ldB = [&](int kk) {
    if (BPATH == 0) {
      // 256 chunks: 8 k-quads x (BNT/G) col-groups; exactly 1 per thread.
      const int g = tid % (BNT / G);
      const int p = tid / (BNT / G);
      const float* base = Bb + (long)(kk + 4 * p) * sBk + (n0 + G * g);
#pragma unroll
      for (int r = 0; r < 4; ++r) {
        if (G == 4) {
          bq[r] = *reinterpret_cast<const float4*>(base + (long)r * sBk);
        } else {
          float2 t = *reinterpret_cast<const float2*>(base + (long)r * sBk);
          bq[r].x = t.x; bq[r].y = t.y;
        }
      }
    } else {
#pragma unroll
      for (int i = 0; i < NB1; ++i) {
        const int e  = tid + i * 256;
        const int n  = e >> 2;
        const int c8 = (e & 3) << 3;
        const float* base = Bb + (long)(n0 + n) * sBn + (kk + c8);
        b1R[i][0] = *reinterpret_cast<const float4*>(base);
        b1R[i][1] = *reinterpret_cast<const float4*>(base + 4);
      }
    }
  };
  auto stB = [&](int buf) {
    if (BPATH == 0) {
      const int g = tid % (BNT / G);
      const int p = tid / (BNT / G);
      if (G == 4) {                        // 16 bf16 col-major quads: 2 x b128
        v8bf lo, hi;
        lo[0] = (__bf16)bq[0].x; lo[1] = (__bf16)bq[1].x;
        lo[2] = (__bf16)bq[2].x; lo[3] = (__bf16)bq[3].x;
        lo[4] = (__bf16)bq[0].y; lo[5] = (__bf16)bq[1].y;
        lo[6] = (__bf16)bq[2].y; lo[7] = (__bf16)bq[3].y;
        hi[0] = (__bf16)bq[0].z; hi[1] = (__bf16)bq[1].z;
        hi[2] = (__bf16)bq[2].z; hi[3] = (__bf16)bq[3].z;
        hi[4] = (__bf16)bq[0].w; hi[5] = (__bf16)bq[1].w;
        hi[6] = (__bf16)bq[2].w; hi[7] = (__bf16)bq[3].w;
        *reinterpret_cast<v8bf*>(&Bs[buf][p][16 * g])     = lo;
        *reinterpret_cast<v8bf*>(&Bs[buf][p][16 * g + 8]) = hi;
      } else {                             // 8 bf16: 1 x b128
        v8bf o;
        o[0] = (__bf16)bq[0].x; o[1] = (__bf16)bq[1].x;
        o[2] = (__bf16)bq[2].x; o[3] = (__bf16)bq[3].x;
        o[4] = (__bf16)bq[0].y; o[5] = (__bf16)bq[1].y;
        o[6] = (__bf16)bq[2].y; o[7] = (__bf16)bq[3].y;
        *reinterpret_cast<v8bf*>(&Bs[buf][p][8 * g]) = o;
      }
    } else {
#pragma unroll
      for (int i = 0; i < NB1; ++i) {
        const int e  = tid + i * 256;
        const int n  = e >> 2;
        const int c8 = (e & 3) << 3;
        v8bf o;
        o[0] = (__bf16)b1R[i][0].x; o[1] = (__bf16)b1R[i][0].y;
        o[2] = (__bf16)b1R[i][0].z; o[3] = (__bf16)b1R[i][0].w;
        o[4] = (__bf16)b1R[i][1].x; o[5] = (__bf16)b1R[i][1].y;
        o[6] = (__bf16)b1R[i][1].z; o[7] = (__bf16)b1R[i][1].w;
        *reinterpret_cast<v8bf*>(&Bs[buf][n][c8]) = o;
      }
    }
  };

  ldA(0); ldB(0);
  stA(0); stB(0);
  __syncthreads();

  const int arow = lane & 15;
  const int akb  = (lane < 16) ? 0 : 16;   // A k base (elements)
  const int bcol = lane & 15;
  const int bkq  = (lane < 16) ? 0 : 4;    // B k-quad row base (path0)
  const int bkb  = (lane < 16) ? 0 : 16;   // B element base    (path1)

  int cur = 0;
  for (int k0 = 0; k0 < K; k0 += BK) {
    const bool nxt = (k0 + BK) < K;
    if (nxt) { ldA(k0 + BK); ldB(k0 + BK); }   // global loads overlap WMMA

    // A fragments: lanes 0-15 row=lane, K=0..15;
    // lanes 16-31 row=lane-16, K=16..31 (matches B). Contiguous 8s -> b128 pairs.
    v16bf afrag[2];
#pragma unroll
    for (int mi = 0; mi < 2; ++mi) {
      const int rb = wm * 32 + mi * 16 + arow;
#pragma unroll
      for (int i = 0; i < 8; ++i) {
        afrag[mi][i]     = As[cur][rb][akb + i];
        afrag[mi][8 + i] = As[cur][rb][akb + 8 + i];
      }
    }
    // B fragments: lanes 0-15 col=lane K=0..15; lanes 16-31 col=lane-16 K=16..31
    v16bf bfrag[NSUB];
#pragma unroll
    for (int ni = 0; ni < NSUB; ++ni) {
      const int cb = wn * (NSUB * 16) + ni * 16 + bcol;
      if (BPATH == 0) {
#pragma unroll
        for (int j = 0; j < 4; ++j) {      // k-quad -> one ds_load_b64 each
          bfrag[ni][4 * j]     = Bs[cur][bkq + j][cb * 4];
          bfrag[ni][4 * j + 1] = Bs[cur][bkq + j][cb * 4 + 1];
          bfrag[ni][4 * j + 2] = Bs[cur][bkq + j][cb * 4 + 2];
          bfrag[ni][4 * j + 3] = Bs[cur][bkq + j][cb * 4 + 3];
        }
      } else {
#pragma unroll
        for (int i = 0; i < 16; ++i) bfrag[ni][i] = Bs[cur][cb][bkb + i];
      }
    }

#pragma unroll
    for (int mi = 0; mi < 2; ++mi)
#pragma unroll
      for (int ni = 0; ni < NSUB; ++ni)
        acc[mi][ni] = __builtin_amdgcn_wmma_f32_16x16x32_bf16(
            false, afrag[mi], false, bfrag[ni], (short)0, acc[mi][ni],
            false, false);

    if (nxt) { stA(cur ^ 1); stB(cur ^ 1); }   // fill other buffer
    __syncthreads();                            // single barrier per k-step
    cur ^= 1;
  }

  // Epilogue: D layout -> m = v + 8*(lane/16), n = lane%16 (exact tiles).
  const int dn  = lane & 15;
  const int dmh = (lane >> 4) * 8;
#pragma unroll
  for (int mi = 0; mi < 2; ++mi) {
#pragma unroll
    for (int ni = 0; ni < NSUB; ++ni) {
      const int gn = n0 + wn * (NSUB * 16) + ni * 16 + dn;
      const float bb = bias ? bias[gn] : 0.f;
#pragma unroll
      for (int vv = 0; vv < 8; ++vv) {
        const int gm = m0 + wm * 32 + mi * 16 + dmh + vv;
        float val = acc[mi][ni][vv] + bb;
        if (relu) val = fmaxf(val, 0.f);
        Cb[(long)gm * ldc + gn] = val;
      }
    }
  }
}

// ---------------------------------------------------------------------------
// Row softmax with scale + optional causal mask. One block (256 thr) per row.
// ---------------------------------------------------------------------------
__global__ __launch_bounds__(256) void softmax_rows(
    float* __restrict__ S, int Tq, int Tk, float scale, int causal)
{
  const long row = blockIdx.x;
  const int  qi  = (int)(row % Tq);
  float* p = S + row * (long)Tk;
  const int t = threadIdx.x;
  __shared__ float red[256];

  float vals[2];
  float m = -1e30f;
#pragma unroll
  for (int i = 0; i < 2; ++i) {
    const int c = t + i * 256;
    float x = p[c] * scale;
    if (causal && c > qi) x = -1e9f;
    vals[i] = x;
    m = fmaxf(m, x);
  }
  red[t] = m; __syncthreads();
  for (int s = 128; s > 0; s >>= 1) {
    if (t < s) red[t] = fmaxf(red[t], red[t + s]);
    __syncthreads();
  }
  m = red[0]; __syncthreads();

  float sum = 0.f;
#pragma unroll
  for (int i = 0; i < 2; ++i) { vals[i] = __expf(vals[i] - m); sum += vals[i]; }
  red[t] = sum; __syncthreads();
  for (int s = 128; s > 0; s >>= 1) {
    if (t < s) red[t] += red[t + s];
    __syncthreads();
  }
  const float inv = 1.f / red[0];
#pragma unroll
  for (int i = 0; i < 2; ++i) {
    const int c = t + i * 256;
    p[c] = vals[i] * inv;
  }
}

// ---------------------------------------------------------------------------
// Fused residual-add + LayerNorm (D=512). In-place safe.
// ---------------------------------------------------------------------------
__global__ __launch_bounds__(256) void layernorm_rows(
    const float* __restrict__ X, const float* __restrict__ R,
    const float* __restrict__ g, const float* __restrict__ b,
    float* __restrict__ out)
{
  const long row = blockIdx.x;
  const int t = threadIdx.x;
  const float* x = X + row * (long)DM;
  const float* r = R ? R + row * (long)DM : nullptr;
  __shared__ float red[256];

  float v[2];
  float s = 0.f;
#pragma unroll
  for (int i = 0; i < 2; ++i) {
    const int c = t + i * 256;
    v[i] = x[c] + (r ? r[c] : 0.f);
    s += v[i];
  }
  red[t] = s; __syncthreads();
  for (int k = 128; k > 0; k >>= 1) { if (t < k) red[t] += red[t + k]; __syncthreads(); }
  const float mean = red[0] * (1.f / DM);
  __syncthreads();

  float s2 = 0.f;
#pragma unroll
  for (int i = 0; i < 2; ++i) { const float d = v[i] - mean; s2 += d * d; }
  red[t] = s2; __syncthreads();
  for (int k = 128; k > 0; k >>= 1) { if (t < k) red[t] += red[t + k]; __syncthreads(); }
  const float inv = rsqrtf(red[0] * (1.f / DM) + 1e-5f);
  __syncthreads();

#pragma unroll
  for (int i = 0; i < 2; ++i) {
    const int c = t + i * 256;
    out[row * (long)DM + c] = (v[i] - mean) * inv * g[c] + b[c];
  }
}

// ---------------------------------------------------------------------------
// Embedding gather + sinusoidal positional encoding.
// ---------------------------------------------------------------------------
__global__ __launch_bounds__(256) void embed_pe(
    const int* __restrict__ tok, const float* __restrict__ emb,
    float* __restrict__ out)
{
  const long row = blockIdx.x;           // b*SEQ + t
  const int  pos = (int)(row % SEQ);
  const int  t   = threadIdx.x;
  const long id  = (long)tok[row];
#pragma unroll
  for (int i = 0; i < 2; ++i) {
    const int c  = t + i * 256;
    const int i2 = c & ~1;
    const float freq = __expf(-9.210340371976184f * (float)i2 / (float)DM);
    const float ang  = (float)pos * freq;
    const float pe   = (c & 1) ? __cosf(ang) : __sinf(ang);
    out[row * (long)DM + c] = emb[id * DM + c] + pe;
  }
}

// ---------------------------------------------------------------------------
// Host-side orchestration
// ---------------------------------------------------------------------------
static inline void launch_gemm(hipStream_t st, const float* A, const float* Bm,
    const float* bias, float* C, int M, int N, int K,
    long lda, long sBk, long sBn, long ldc,
    int batch, int Hb,
    long oAb, long oAh, long oBb, long oBh, long oCb, long oCh, int relu)
{
  if (N % 128 == 0) {
    dim3 grid(N / 128, (M + BM - 1) / BM, batch);
    if (sBn == 1)
      gemm_wmma_bf16<0, 128><<<grid, 256, 0, st>>>(A, Bm, bias, C, M, N, K,
          lda, sBk, sBn, ldc, Hb, oAb, oAh, oBb, oBh, oCb, oCh, relu);
    else
      gemm_wmma_bf16<1, 128><<<grid, 256, 0, st>>>(A, Bm, bias, C, M, N, K,
          lda, sBk, sBn, ldc, Hb, oAb, oAh, oBb, oBh, oCb, oCh, relu);
  } else {
    dim3 grid((N + 63) / 64, (M + BM - 1) / BM, batch);
    if (sBn == 1)
      gemm_wmma_bf16<0, 64><<<grid, 256, 0, st>>>(A, Bm, bias, C, M, N, K,
          lda, sBk, sBn, ldc, Hb, oAb, oAh, oBb, oBh, oCb, oCh, relu);
    else
      gemm_wmma_bf16<1, 64><<<grid, 256, 0, st>>>(A, Bm, bias, C, M, N, K,
          lda, sBk, sBn, ldc, Hb, oAb, oAh, oBb, oBh, oCb, oCh, relu);
  }
}

// Multi-head attention: proj = O-proj( softmax(Q K^T * s [+mask]) V )
static void run_mha(hipStream_t st,
    const float* qin, const float* kvin,
    const float* qw, const float* qb, const float* kw, const float* kb,
    const float* vw, const float* vb, const float* ow, const float* ob,
    float* q, float* k, float* v, float* scores, float* ctx, float* proj,
    int causal)
{
  const int MR = BATCH * SEQ;
  launch_gemm(st, qin,  qw, qb, q, MR, DM, DM, DM, DM, 1, DM, 1, 1, 0,0,0,0,0,0, 0);
  launch_gemm(st, kvin, kw, kb, k, MR, DM, DM, DM, DM, 1, DM, 1, 1, 0,0,0,0,0,0, 0);
  launch_gemm(st, kvin, vw, vb, v, MR, DM, DM, DM, DM, 1, DM, 1, 1, 0,0,0,0,0,0, 0);
  // scores[b,h] = Q[b,:,hHD:] @ K[b,:,hHD:]^T  (sBk==1 -> transposed LDS path)
  launch_gemm(st, q, k, nullptr, scores, SEQ, SEQ, HD,
      DM, 1, DM, SEQ,
      BATCH * NH, NH,
      (long)SEQ * DM, HD, (long)SEQ * DM, HD,
      (long)NH * SEQ * SEQ, (long)SEQ * SEQ, 0);
  softmax_rows<<<BATCH * NH * SEQ, 256, 0, st>>>(scores, SEQ, SEQ, ATT_SCALE, causal);
  // ctx[b,:,hHD:] = P[b,h] @ V[b,:,hHD:]   (N=64 -> BNT=64 instantiation)
  launch_gemm(st, scores, v, nullptr, ctx, SEQ, HD, SEQ,
      SEQ, DM, 1, DM,
      BATCH * NH, NH,
      (long)NH * SEQ * SEQ, (long)SEQ * SEQ,
      (long)SEQ * DM, HD,
      (long)SEQ * DM, HD, 0);
  launch_gemm(st, ctx, ow, ob, proj, MR, DM, DM, DM, DM, 1, DM, 1, 1, 0,0,0,0,0,0, 0);
}

extern "C" void kernel_launch(void* const* d_in, const int* in_sizes, int n_in,
                              void* d_out, int out_size, void* d_ws, size_t ws_size,
                              hipStream_t stream)
{
  // Flat input order follows setup_inputs() dict insertion order (recursive).
  const float* src_emb = (const float*)d_in[0];
  const float* tgt_emb = (const float*)d_in[1];
  const float* e_qw = (const float*)d_in[2];  const float* e_qb = (const float*)d_in[3];
  const float* e_kw = (const float*)d_in[4];  const float* e_kb = (const float*)d_in[5];
  const float* e_vw = (const float*)d_in[6];  const float* e_vb = (const float*)d_in[7];
  const float* e_ow = (const float*)d_in[8];  const float* e_ob = (const float*)d_in[9];
  const float* e_w1 = (const float*)d_in[10]; const float* e_b1 = (const float*)d_in[11];
  const float* e_w2 = (const float*)d_in[12]; const float* e_b2 = (const float*)d_in[13];
  const float* e_l1g = (const float*)d_in[14]; const float* e_l1b = (const float*)d_in[15];
  const float* e_l2g = (const float*)d_in[16]; const float* e_l2b = (const float*)d_in[17];
  const float* d_qw = (const float*)d_in[18]; const float* d_qb = (const float*)d_in[19];
  const float* d_kw = (const float*)d_in[20]; const float* d_kb = (const float*)d_in[21];
  const float* d_vw = (const float*)d_in[22]; const float* d_vb = (const float*)d_in[23];
  const float* d_ow = (const float*)d_in[24]; const float* d_ob = (const float*)d_in[25];
  const float* c_qw = (const float*)d_in[26]; const float* c_qb = (const float*)d_in[27];
  const float* c_kw = (const float*)d_in[28]; const float* c_kb = (const float*)d_in[29];
  const float* c_vw = (const float*)d_in[30]; const float* c_vb = (const float*)d_in[31];
  const float* c_ow = (const float*)d_in[32]; const float* c_ob = (const float*)d_in[33];
  const float* d_w1 = (const float*)d_in[34]; const float* d_b1 = (const float*)d_in[35];
  const float* d_w2 = (const float*)d_in[36]; const float* d_b2 = (const float*)d_in[37];
  const float* d_l1g = (const float*)d_in[38]; const float* d_l1b = (const float*)d_in[39];
  const float* d_l2g = (const float*)d_in[40]; const float* d_l2b = (const float*)d_in[41];
  const float* d_l3g = (const float*)d_in[42]; const float* d_l3b = (const float*)d_in[43];
  const float* enc_lng = (const float*)d_in[44]; const float* enc_lnb = (const float*)d_in[45];
  const float* dec_lng = (const float*)d_in[46]; const float* dec_lnb = (const float*)d_in[47];
  const float* out_w   = (const float*)d_in[48]; const float* out_b   = (const float*)d_in[49];
  const int* src_tok = (const int*)d_in[50];
  const int* tgt_tok = (const int*)d_in[51];
  float* logits = (float*)d_out;

  // Workspace layout (floats). Total ~20.5M floats (~82 MB).
  float* ws = (float*)d_ws;
  const size_t BTD = (size_t)BATCH * SEQ * DM;       // 1,048,576
  float* x      = ws;
  float* y      = x + BTD;
  float* q      = y + BTD;
  float* k      = q + BTD;
  float* v      = k + BTD;
  float* ctx    = v + BTD;
  float* proj   = ctx + BTD;
  float* mem    = proj + BTD;
  float* ffn    = mem + BTD;                         // [B,T,F]
  float* scores = ffn + (size_t)BATCH * SEQ * FF;    // [B,H,T,T]

  const int MR = BATCH * SEQ;
  const long DD = (long)DM * DM;

  // ---------------- Encoder ----------------
  embed_pe<<<MR, 256, 0, stream>>>(src_tok, src_emb, x);
  for (int l = 0; l < NL; ++l) {
    run_mha(stream, x, x,
        e_qw + l * DD, e_qb + l * DM, e_kw + l * DD, e_kb + l * DM,
        e_vw + l * DD, e_vb + l * DM, e_ow + l * DD, e_ob + l * DM,
        q, k, v, scores, ctx, proj, /*causal=*/0);
    layernorm_rows<<<MR, 256, 0, stream>>>(x, proj, e_l1g + l * DM, e_l1b + l * DM, x);
    launch_gemm(stream, x, e_w1 + (long)l * DM * FF, e_b1 + (long)l * FF, ffn,
        MR, FF, DM, DM, FF, 1, FF, 1, 1, 0,0,0,0,0,0, /*relu=*/1);
    launch_gemm(stream, ffn, e_w2 + (long)l * FF * DM, e_b2 + l * DM, proj,
        MR, DM, FF, FF, DM, 1, DM, 1, 1, 0,0,0,0,0,0, 0);
    layernorm_rows<<<MR, 256, 0, stream>>>(x, proj, e_l2g + l * DM, e_l2b + l * DM, x);
  }
  layernorm_rows<<<MR, 256, 0, stream>>>(x, nullptr, enc_lng, enc_lnb, mem);

  // ---------------- Decoder ----------------
  embed_pe<<<MR, 256, 0, stream>>>(tgt_tok, tgt_emb, y);
  for (int l = 0; l < NL; ++l) {
    run_mha(stream, y, y,
        d_qw + l * DD, d_qb + l * DM, d_kw + l * DD, d_kb + l * DM,
        d_vw + l * DD, d_vb + l * DM, d_ow + l * DD, d_ob + l * DM,
        q, k, v, scores, ctx, proj, /*causal=*/1);
    layernorm_rows<<<MR, 256, 0, stream>>>(y, proj, d_l1g + l * DM, d_l1b + l * DM, y);
    run_mha(stream, y, mem,
        c_qw + l * DD, c_qb + l * DM, c_kw + l * DD, c_kb + l * DM,
        c_vw + l * DD, c_vb + l * DM, c_ow + l * DD, c_ob + l * DM,
        q, k, v, scores, ctx, proj, /*causal=*/0);
    layernorm_rows<<<MR, 256, 0, stream>>>(y, proj, d_l2g + l * DM, d_l2b + l * DM, y);
    launch_gemm(stream, y, d_w1 + (long)l * DM * FF, d_b1 + (long)l * FF, ffn,
        MR, FF, DM, DM, FF, 1, FF, 1, 1, 0,0,0,0,0,0, /*relu=*/1);
    launch_gemm(stream, ffn, d_w2 + (long)l * FF * DM, d_b2 + l * DM, proj,
        MR, DM, FF, FF, DM, 1, DM, 1, 1, 0,0,0,0,0,0, 0);
    layernorm_rows<<<MR, 256, 0, stream>>>(y, proj, d_l3g + l * DM, d_l3b + l * DM, y);
  }
  layernorm_rows<<<MR, 256, 0, stream>>>(y, nullptr, dec_lng, dec_lnb, y);

  // ---------------- Output head: [2048,512] @ [512,32000] + bias -----------
  launch_gemm(stream, y, out_w, out_b, logits, MR, VOC, DM,
      DM, VOC, 1, VOC, 1, 1, 0,0,0,0,0,0, 0);
}
